// MultiHeadSelfAttention_59365037965486
// MI455X (gfx1250) — matrix operand order repaired and checked by reading, not compiled, not run on hardware
//
#include <hip/hip_runtime.h>
#include <hip/hip_bf16.h>

// MI455X multi-head self-attention, WMMA f16 (f32 accum) end-to-end.
// B=4 T=2048 C=1024 H=16 D=64. Compute-bound (~120 GFLOP vs ~100MB traffic).
// v3: async global->LDS double-buffered tiles + raw split barriers (no fence
//     counter-drain, so register prefetches survive across barriers) +
//     one-stage pipelined LDS fragment reads for partial s_wait_dscnt.

typedef __attribute__((ext_vector_type(16))) _Float16 v16h;
typedef __attribute__((ext_vector_type(8)))  _Float16 v8h;
typedef __attribute__((ext_vector_type(4)))  _Float16 v4h;
typedef __attribute__((ext_vector_type(8)))  float    v8f;
typedef __attribute__((ext_vector_type(4)))  float    v4f;

union U16 { v16h v; v8h h[2]; };

static __device__ __forceinline__ v8f wmma16(v16h a, v16h b, v8f c) {
  return __builtin_amdgcn_wmma_f32_16x16x32_f16(false, a, false, b, (short)0, c,
                                                false, false);
}

// ---- CDNA5 async global -> LDS copy (16B per lane), tracked by ASYNCcnt ----
typedef __attribute__((address_space(3))) _Float16 lds_f16;
static __device__ __forceinline__ void cp_g2l_b128(_Float16* lds_dst,
                                                   const _Float16* gsrc) {
  unsigned loff = (unsigned)(size_t)(lds_f16*)lds_dst;  // LDS byte address
  unsigned long long ga = (unsigned long long)(size_t)gsrc;
  asm volatile("global_load_async_to_lds_b128 %0, %1, off"
               :: "v"(loff), "v"(ga) : "memory");
}
static __device__ __forceinline__ void wait_async0() {
  asm volatile("s_wait_asynccnt 0x0" ::: "memory");
}
// Split workgroup barrier WITHOUT the seq-cst fence __syncthreads() implies:
// async-DMA LDS writes are visible once ASYNCcnt drains, so no vmem drain
// is needed and register prefetches stay in flight across the barrier.
static __device__ __forceinline__ void wg_barrier() {
  asm volatile("s_barrier_signal -1\n\ts_barrier_wait -1" ::: "memory");
}

#define TT 2048
#define CC 1024
#define HH 16
#define DD 64

// ---------------------------------------------------------------- f32 -> f16
__global__ __launch_bounds__(256) void cvt_f32_f16(const float* __restrict__ in,
                                                   _Float16* __restrict__ out,
                                                   int n4) {
  int i = blockIdx.x * 256 + threadIdx.x;
  if (i < n4) {
    v4f v = ((const v4f*)in)[i];
    v4h o;
    o[0] = (_Float16)v[0]; o[1] = (_Float16)v[1];
    o[2] = (_Float16)v[2]; o[3] = (_Float16)v[3];
    ((v4h*)out)[i] = o;
  }
}

// -------------------------------------- W[K][N] f32 -> Wt[N][K] f16 (tiled)
__global__ __launch_bounds__(256) void transpose_w(const float* __restrict__ W,
                                                   _Float16* __restrict__ Wt) {
  __shared__ _Float16 tile[32][33];
  const int bx = blockIdx.x * 32;  // n base
  const int by = blockIdx.y * 32;  // k base
  const int tx = threadIdx.x & 31;
  const int ty = threadIdx.x >> 5;  // 8 rows per pass
#pragma unroll
  for (int r = 0; r < 32; r += 8)
    tile[ty + r][tx] = (_Float16)W[(size_t)(by + ty + r) * CC + bx + tx];
  __syncthreads();
#pragma unroll
  for (int r = 0; r < 32; r += 8)
    Wt[(size_t)(bx + ty + r) * CC + by + tx] = tile[tx][ty + r];
}

// ------------------------------------------------------------- WMMA GEMM ---
// C[M x 1024] = A[M x 1024] * Bt^T + bias. 8 waves; wave -> 16 rows x 128 cols.
// B tile (128n x 32k, 8KB) async-staged to LDS, double-buffered; A frags
// prefetched one k-step ahead in registers; B frags read one wmma ahead.
// mode 0: Q->f16 [B,H,T,D] scaled; 1: K->f16 [B,H,T,D]; 2: V->f16 [B,H,D,T];
// mode 3: out->f32 [B,T,C]
__global__ __launch_bounds__(256) void gemm_wmma(
    const _Float16* __restrict__ A, const _Float16* __restrict__ Bt,
    const float* __restrict__ bias, void* __restrict__ outv, int mode,
    float scale) {
  __shared__ __align__(16) _Float16 bbuf[2][128 * 32];
  const int tid  = threadIdx.x;
  const int lane = tid & 31;
  const int wave = tid >> 5;
  const int g  = lane >> 4;
  const int ln = lane & 15;
  const int mbase = blockIdx.x * 128 + wave * 16;
  const int nbase = blockIdx.y * 128;

  v8f acc[8] = {};
  const _Float16* aptr = A + (size_t)(mbase + ln) * CC + g * 16;
  const _Float16* bgp  = Bt + (size_t)(nbase + (tid >> 1)) * CC + (tid & 1) * 16;

  // stage B tile for kb into LDS buffer `buf`: bbuf[buf][n_local*32 + k_local]
  auto stage_B = [&](int kb, int buf) {
    _Float16* dst = &bbuf[buf][(tid >> 1) * 32 + (tid & 1) * 16];
    cp_g2l_b128(dst,     bgp + kb);
    cp_g2l_b128(dst + 8, bgp + kb + 8);
  };

  stage_B(0, 0);
  U16 a;
  a.h[0] = *(const v8h*)(aptr);
  a.h[1] = *(const v8h*)(aptr + 8);
  wait_async0();
  wg_barrier();

  for (int kb = 0; kb < CC; kb += 32) {
    const int cur = (kb >> 5) & 1;
    if (kb + 32 < CC) stage_B(kb + 32, cur ^ 1);   // async fill of other buffer
    U16 an = a;
    if (kb + 32 < CC) {                            // prefetch next A frag
      an.h[0] = *(const v8h*)(aptr + kb + 32);
      an.h[1] = *(const v8h*)(aptr + kb + 40);
    }
    // one-stage pipelined LDS reads: load frag f+1 before wmma f
    v16h bcur = *(const v16h*)&bbuf[cur][ln * 32 + g * 16];
#pragma unroll
    for (int f = 0; f < 8; ++f) {
      v16h bnext = bcur;
      if (f < 7)
        bnext = *(const v16h*)&bbuf[cur][((f + 1) * 16 + ln) * 32 + g * 16];
      acc[f] = wmma16(a.v, bcur, acc[f]);
      bcur = bnext;
    }
    a = an;
    wait_async0();
    wg_barrier();
  }

#pragma unroll
  for (int f = 0; f < 8; ++f) {
    const int n = nbase + f * 16 + ln;   // C frag: col = ln, row = r + 8*g
    const float bs = bias[n];
    if (mode == 3) {
      float* O = (float*)outv;
#pragma unroll
      for (int r = 0; r < 8; ++r) {
        const int m = mbase + r + 8 * g;
        O[(size_t)m * CC + n] = (acc[f][r] + bs) * scale;
      }
    } else {
      _Float16* O = (_Float16*)outv;
      const int h = n >> 6, d = n & 63;
      if (mode == 2) {  // Vt[(b*H+h)][d][t] : t contiguous over r -> b128 store
        const int m0 = mbase + 8 * g;
        const int bb = m0 >> 11, t0 = m0 & (TT - 1);
        v8h pk;
#pragma unroll
        for (int r = 0; r < 8; ++r) pk[r] = (_Float16)((acc[f][r] + bs) * scale);
        *(v8h*)(O + ((size_t)(bb * HH + h) * DD + d) * TT + t0) = pk;
      } else {  // Q/K [(b*H+h)][t][d]
#pragma unroll
        for (int r = 0; r < 8; ++r) {
          const int m = mbase + r + 8 * g;
          const int bb = m >> 11, t = m & (TT - 1);
          O[((size_t)(bb * HH + h) * TT + t) * DD + d] =
              (_Float16)((acc[f][r] + bs) * scale);
        }
      }
    }
  }
}

// -------------------------------------------------- flash attention (causal)
// grid (T/128, B*H); 8 waves each own 16 query rows. K/V blocks of 32 keys
// async-staged to LDS (shared by all 8 waves), double-buffered; per step:
// QK^T (4 wmma) -> online softmax -> P relayout via per-wave LDS -> PV (4 wmma).
__global__ __launch_bounds__(256) void attn_wmma(
    const _Float16* __restrict__ Q, const _Float16* __restrict__ K,
    const _Float16* __restrict__ Vt, _Float16* __restrict__ Y) {
  __shared__ __align__(16) _Float16 kbuf[2][32 * 64];  // [key][d]
  __shared__ __align__(16) _Float16 vbuf[2][64 * 32];  // [d][key]
  __shared__ __align__(16) _Float16 pbuf[8 * 512];
  const int tid  = threadIdx.x;
  const int lane = tid & 31;
  const int wave = tid >> 5;
  const int g  = lane >> 4;
  const int ln = lane & 15;
  const int bh = blockIdx.y;
  const int bb = bh >> 4, h = bh & 15;
  const int tbase = blockIdx.x * 128 + wave * 16;

  const _Float16* Qb = Q + (size_t)bh * TT * DD;
  const _Float16* Kb = K + (size_t)bh * TT * DD;
  const _Float16* Vb = Vt + (size_t)bh * DD * TT;
  _Float16* pw = pbuf + wave * 512;

  // stage one 32-key block: K rows (32x64) + Vt cols (64x32); 2 b128/thread
  auto stage_kv = [&](int sb, int buf) {
    cp_g2l_b128(&kbuf[buf][(tid >> 3) * 64 + (tid & 7) * 8],
                Kb + (size_t)(sb + (tid >> 3)) * DD + (tid & 7) * 8);
    cp_g2l_b128(&vbuf[buf][(tid >> 2) * 32 + (tid & 3) * 8],
                Vb + (size_t)(tid >> 2) * TT + sb + (tid & 3) * 8);
  };

  // Q A-frags: 16 rows x 64 d, two K=32 fragments (d 0..31 and 32..63)
  U16 qa0, qa1;
  {
    const _Float16* qp = Qb + (size_t)(tbase + ln) * DD + g * 16;
    qa0.h[0] = *(const v8h*)(qp);
    qa0.h[1] = *(const v8h*)(qp + 8);
    qa1.h[0] = *(const v8h*)(qp + 32);
    qa1.h[1] = *(const v8h*)(qp + 40);
  }

  v8f o[4] = {};
  float mrow[8], lrow[8];
#pragma unroll
  for (int r = 0; r < 8; ++r) { mrow[r] = -3.0e38f; lrow[r] = 0.f; }

  const int tmax = tbase + 15;                   // this wave's causal bound
  const int block_tmax = blockIdx.x * 128 + 127; // block's causal bound

  stage_kv(0, 0);
  wait_async0();
  wg_barrier();

  for (int sb = 0; sb <= block_tmax; sb += 32) {
    const int cur = (sb >> 5) & 1;
    if (sb + 32 <= block_tmax) stage_kv(sb + 32, cur ^ 1);
    if (sb <= tmax) {  // wave-uniform predicate; barriers stay outside
      // ---- scores: S[16 x 32] = Q(16x64) . K^T  (B frags from LDS)
      v16h kf[4];
#pragma unroll
      for (int sub = 0; sub < 2; ++sub)
#pragma unroll
        for (int dp = 0; dp < 2; ++dp)
          kf[sub * 2 + dp] =
              *(const v16h*)&kbuf[cur][(sub * 16 + ln) * 64 + dp * 32 + g * 16];
      v8f s0 = {}, s1 = {};
      // interleave the two accumulator chains for ILP
      s0 = wmma16(qa0.v, kf[0], s0);
      s1 = wmma16(qa0.v, kf[2], s1);
      s0 = wmma16(qa1.v, kf[1], s0);
      s1 = wmma16(qa1.v, kf[3], s1);
      // ---- causal mask (only near the diagonal)
      if (sb + 31 > tbase) {
#pragma unroll
        for (int r = 0; r < 8; ++r) {
          const int t = tbase + r + 8 * g;
          if (sb + ln > t)      s0[r] = -3.0e38f;
          if (sb + 16 + ln > t) s1[r] = -3.0e38f;
        }
      }
      // ---- online softmax; a row spans the 16 lanes of each half-group
      float rmax[8];
#pragma unroll
      for (int r = 0; r < 8; ++r) rmax[r] = fmaxf(s0[r], s1[r]);
#pragma unroll
      for (int off = 1; off < 16; off <<= 1)
#pragma unroll
        for (int r = 0; r < 8; ++r)
          rmax[r] = fmaxf(rmax[r], __shfl_xor(rmax[r], off, 32));
      float p0[8], p1[8], al[8], rs[8];
#pragma unroll
      for (int r = 0; r < 8; ++r) {
        const float mn = fmaxf(mrow[r], rmax[r]);
        al[r] = __expf(mrow[r] - mn);
        p0[r] = __expf(s0[r] - mn);
        p1[r] = __expf(s1[r] - mn);
        mrow[r] = mn;
        rs[r] = p0[r] + p1[r];
      }
#pragma unroll
      for (int off = 1; off < 16; off <<= 1)
#pragma unroll
        for (int r = 0; r < 8; ++r) rs[r] += __shfl_xor(rs[r], off, 32);
#pragma unroll
      for (int r = 0; r < 8; ++r) lrow[r] = lrow[r] * al[r] + rs[r];
#pragma unroll
      for (int f = 0; f < 4; ++f)
#pragma unroll
        for (int r = 0; r < 8; ++r) o[f][r] *= al[r];
      // ---- C-frag -> A-frag relayout through per-wave LDS (DS in-order)
#pragma unroll
      for (int r = 0; r < 8; ++r) {
        pw[(r + 8 * g) * 32 + ln]      = (_Float16)p0[r];
        pw[(r + 8 * g) * 32 + 16 + ln] = (_Float16)p1[r];
      }
      U16 pa;
      pa.h[0] = *(const v8h*)(pw + ln * 32 + g * 16);
      pa.h[1] = *(const v8h*)(pw + ln * 32 + g * 16 + 8);
      // ---- O += P(16x32) . V(32x64); pipelined V-frag reads
      v16h vcur = *(const v16h*)&vbuf[cur][ln * 32 + g * 16];
#pragma unroll
      for (int f = 0; f < 4; ++f) {
        v16h vnext = vcur;
        if (f < 3)
          vnext = *(const v16h*)&vbuf[cur][((f + 1) * 16 + ln) * 32 + g * 16];
        o[f] = wmma16(pa.v, vcur, o[f]);
        vcur = vnext;
      }
    }
    wait_async0();
    wg_barrier();
  }

  float inv[8];
#pragma unroll
  for (int r = 0; r < 8; ++r) inv[r] = 1.0f / lrow[r];
#pragma unroll
  for (int f = 0; f < 4; ++f)
#pragma unroll
    for (int r = 0; r < 8; ++r) {
      const int t = tbase + r + 8 * g;
      Y[((size_t)(bb * TT + t)) * CC + h * DD + f * 16 + ln] =
          (_Float16)(o[f][r] * inv[r]);
    }
}

// ---------------------------------------------------------------------------
extern "C" void kernel_launch(void* const* d_in, const int* in_sizes, int n_in,
                              void* d_out, int out_size, void* d_ws,
                              size_t ws_size, hipStream_t stream) {
  const float* x  = (const float*)d_in[0];
  const float* Wq = (const float*)d_in[1];
  const float* bq = (const float*)d_in[2];
  const float* Wk = (const float*)d_in[3];
  const float* bk = (const float*)d_in[4];
  const float* Wv = (const float*)d_in[5];
  const float* bv = (const float*)d_in[6];
  const float* Wp = (const float*)d_in[7];
  const float* bp = (const float*)d_in[8];

  const size_t M = 4ull * TT;            // 8192 rows
  const size_t SZ_X = M * CC;            // 8,388,608 halfs
  const size_t SZ_W = (size_t)CC * CC;   // 1,048,576 halfs

  _Float16* ws  = (_Float16*)d_ws;
  _Float16* XH  = ws;
  _Float16* WQT = XH + SZ_X;
  _Float16* WKT = WQT + SZ_W;
  _Float16* WVT = WKT + SZ_W;
  _Float16* WPT = WVT + SZ_W;
  _Float16* QH  = WPT + SZ_W;
  _Float16* KH  = QH + SZ_X;
  _Float16* VTH = KH + SZ_X;
  _Float16* YH  = VTH + SZ_X;

  // stage 0: precision conversion + weight transpose (one-time, ~24MB)
  cvt_f32_f16<<<(int)(SZ_X / 4 / 256), 256, 0, stream>>>(x, XH, (int)(SZ_X / 4));
  transpose_w<<<dim3(32, 32), 256, 0, stream>>>(Wq, WQT);
  transpose_w<<<dim3(32, 32), 256, 0, stream>>>(Wk, WKT);
  transpose_w<<<dim3(32, 32), 256, 0, stream>>>(Wv, WVT);
  transpose_w<<<dim3(32, 32), 256, 0, stream>>>(Wp, WPT);

  // stage 1: QKV projections (scale 1/sqrt(64) folded into Q)
  const dim3 ggrid((unsigned)(M / 128), CC / 128);
  gemm_wmma<<<ggrid, 256, 0, stream>>>(XH, WQT, bq, QH, 0, 0.125f);
  gemm_wmma<<<ggrid, 256, 0, stream>>>(XH, WKT, bk, KH, 1, 1.0f);
  gemm_wmma<<<ggrid, 256, 0, stream>>>(XH, WVT, bv, VTH, 2, 1.0f);

  // stage 2: causal flash attention
  attn_wmma<<<dim3(TT / 128, 4 * HH), 256, 0, stream>>>(QH, KH, VTH, YH);

  // stage 3: output projection -> f32
  gemm_wmma<<<ggrid, 256, 0, stream>>>(YH, WPT, bp, d_out, 3, 1.0f);
}
